// CrossAttention_82325933130036
// MI455X (gfx1250) — matrix operand order repaired and checked by reading, not compiled, not run on hardware
//
#include <hip/hip_runtime.h>
#include <hip/hip_bf16.h>

// ---------------------------------------------------------------------------
// CDNA5 (gfx1250) fused cross-attention: bf16 WMMA + flash streaming with
// GLOBAL_LOAD_ASYNC_TO_LDS (ASYNCcnt) tile staging.
// ---------------------------------------------------------------------------

typedef __attribute__((ext_vector_type(8)))  float  v8f;
typedef __attribute__((ext_vector_type(16))) __bf16 v16bf;
typedef __attribute__((ext_vector_type(2)))  __bf16 v2bf;

union BF16Frag {
    uint4  q[2];   // two 16-byte chunks, element order = VGPR order
    v16bf  v;
};

// Problem constants (reference: B=16, Nq=Nk=2048, F=512, D=256)
#define BB   16
#define NQ   2048
#define NK   2048
#define DD   256      // qk dim
#define FF   512      // feature dim
#define RR   (BB*NQ)  // 32768 rows for projections

// ---------------- f32 -> bf16 (RNE) ----------------
__device__ __forceinline__ unsigned short f2bf(float f) {
    unsigned u = __builtin_bit_cast(unsigned, f);
    unsigned r = u + 0x7FFFu + ((u >> 16) & 1u);
    return (unsigned short)(r >> 16);
}

#if defined(__has_builtin)
#if __has_builtin(__builtin_amdgcn_cvt_pk_bf16_f32)
#define HAVE_CVT_PK_BF16 1
#endif
#endif

__device__ __forceinline__ unsigned f2bf2(float a, float b) {
#ifdef HAVE_CVT_PK_BF16
    v2bf p = __builtin_amdgcn_cvt_pk_bf16_f32(a, b);   // v_cvt_pk_bf16_f32
    return __builtin_bit_cast(unsigned, p);
#else
    return (unsigned)f2bf(a) | ((unsigned)f2bf(b) << 16);
#endif
}

__device__ __forceinline__ v8f wmma_bf16(const BF16Frag& a, const BF16Frag& b, v8f c) {
    return __builtin_amdgcn_wmma_f32_16x16x32_bf16(
        /*neg_a=*/false, a.v, /*neg_b=*/false, b.v,
        /*c_mod=*/(short)0, c, /*reuse_a=*/false, /*reuse_b=*/false);
}

// ---------------- CDNA5 async memory->LDS (inline asm, ASYNCcnt) ----------------
typedef __attribute__((address_space(3))) char lds_char_t;

__device__ __forceinline__ unsigned lds_off(const void* p) {
    return (unsigned)(unsigned long long)(lds_char_t*)p;   // addrspacecast -> 32-bit LDS offset
}
// GVS mode: mem = SGPR_base + VGPR_offset; LDS[vdst] = MEM[mem] (16 bytes/lane)
__device__ __forceinline__ void async_ld16(void* lds_p, const void* sbase, unsigned voff) {
    unsigned l = lds_off(lds_p);
    asm volatile("global_load_async_to_lds_b128 %0, %1, %2"
                 :
                 : "v"(l), "v"(voff), "s"(sbase)
                 : "memory");
}
__device__ __forceinline__ void wait_async_le16() {
    asm volatile("s_wait_asynccnt 0x10" ::: "memory");
}
__device__ __forceinline__ void wait_async_0() {
    asm volatile("s_wait_asynccnt 0x0" ::: "memory");
}

// ---------------------------------------------------------------------------
// Projection GEMM:  Y[r,n] = sum_k X[r,k] * W[n,k]   (f32 in, f32 out)
// WG = 256 threads (8 waves). Tile: 128 rows x 64 cols, K-step 32.
// ---------------------------------------------------------------------------
#define GXS 40   // bf16 elems per LDS row (32 + 8 pad) -> 80B, 16B-aligned rows

__global__ void __launch_bounds__(256, 2)
proj_gemm_kernel(const float* __restrict__ X, const float* __restrict__ W,
                 float* __restrict__ Y, int K, int N)
{
    __shared__ unsigned short Xs[128 * GXS];
    __shared__ unsigned short Ws[64 * GXS];

    const int t    = threadIdx.x;
    const int lane = t & 31;
    const int wave = t >> 5;
    const int lo   = lane & 15;
    const int hi   = lane >> 4;
    const long rowbase = (long)blockIdx.x * 128;
    const int  n0      = blockIdx.y * 64;

    const v8f vzero = {0.f,0.f,0.f,0.f,0.f,0.f,0.f,0.f};
    v8f acc[4];
#pragma unroll
    for (int f = 0; f < 4; ++f) acc[f] = vzero;

    for (int kc = 0; kc < K; kc += 32) {
        // stage X tile 128x32 f32 -> bf16 LDS
#pragma unroll
        for (int i = 0; i < 4; ++i) {
            int idx = t + i * 256;
            int row = idx >> 3, c4 = idx & 7;
            float4 v = *(const float4*)(X + (rowbase + row) * (long)K + kc + c4 * 4);
            uint2 p; p.x = f2bf2(v.x, v.y); p.y = f2bf2(v.z, v.w);
            *(uint2*)(Xs + row * GXS + c4 * 4) = p;
        }
        // stage W tile 64x32 f32 -> bf16 LDS
#pragma unroll
        for (int i = 0; i < 2; ++i) {
            int idx = t + i * 256;
            int row = idx >> 3, c4 = idx & 7;
            float4 v = *(const float4*)(W + (long)(n0 + row) * K + kc + c4 * 4);
            uint2 p; p.x = f2bf2(v.x, v.y); p.y = f2bf2(v.z, v.w);
            *(uint2*)(Ws + row * GXS + c4 * 4) = p;
        }
        __syncthreads();

        BF16Frag a;   // A: 16x32, per-lane chunks at K-offsets hi*8 and hi*8+16
        {
            const char* ab = (const char*)Xs + ((wave * 16 + lo) * GXS + hi * 8) * 2;
            a.q[0] = *(const uint4*)ab;
            a.q[1] = *(const uint4*)(ab + 32);
        }
#pragma unroll
        for (int f = 0; f < 4; ++f) {
            BF16Frag b;   // B: 32x16, lane n=lo -> W row f*16+lo, chunks at hi*8 and hi*8+16
            const char* bb = (const char*)Ws + ((f * 16 + lo) * GXS + hi * 8) * 2;
            b.q[0] = *(const uint4*)bb;
            b.q[1] = *(const uint4*)(bb + 32);
            acc[f] = wmma_bf16(a, b, acc[f]);
        }
        __syncthreads();
    }

    const long row0 = rowbase + wave * 16 + hi * 8;
#pragma unroll
    for (int f = 0; f < 4; ++f)
#pragma unroll
        for (int r = 0; r < 8; ++r)
            Y[(row0 + r) * (long)N + n0 + f * 16 + lo] = acc[f][r];
}

// ---------------------------------------------------------------------------
// Row L2-normalize f32 -> bf16, row-major. One wave per row.
// ---------------------------------------------------------------------------
__global__ void __launch_bounds__(256)
rownorm_kernel(const float* __restrict__ Y, unsigned short* __restrict__ O, int C)
{
    const int row  = blockIdx.x * 8 + (threadIdx.x >> 5);
    const int lane = threadIdx.x & 31;
    const int chunk = C >> 5;
    const float* p = Y + (long)row * C + lane * chunk;
    float ss = 0.f;
    for (int j = 0; j < chunk; j += 4) {
        float4 v = *(const float4*)(p + j);
        ss += v.x*v.x + v.y*v.y + v.z*v.z + v.w*v.w;
    }
#pragma unroll
    for (int m = 16; m; m >>= 1) ss += __shfl_xor(ss, m, 32);
    const float inv = 1.0f / fmaxf(sqrtf(ss), 1e-12f);
    unsigned short* q = O + (long)row * C + lane * chunk;
    for (int j = 0; j < chunk; j += 4) {
        float4 v = *(const float4*)(p + j);
        uint2 u; u.x = f2bf2(v.x * inv, v.y * inv); u.y = f2bf2(v.z * inv, v.w * inv);
        *(uint2*)(q + j) = u;
    }
}

// ---------------------------------------------------------------------------
// Row L2-normalize f32 -> bf16, TRANSPOSED output OT[b][v][m] (m contiguous).
// ---------------------------------------------------------------------------
__global__ void __launch_bounds__(256)
rownorm_T_kernel(const float* __restrict__ Y, unsigned short* __restrict__ OT)
{
    const int r    = blockIdx.x * 8 + (threadIdx.x >> 5);
    const int lane = threadIdx.x & 31;
    const int b = r >> 11, m = r & 2047;
    const float* p = Y + (long)r * FF + lane * 16;
    float ss = 0.f;
#pragma unroll
    for (int j = 0; j < 16; j += 4) {
        float4 v = *(const float4*)(p + j);
        ss += v.x*v.x + v.y*v.y + v.z*v.z + v.w*v.w;
    }
#pragma unroll
    for (int mm = 16; mm; mm >>= 1) ss += __shfl_xor(ss, mm, 32);
    const float inv = 1.0f / fmaxf(sqrtf(ss), 1e-12f);
#pragma unroll
    for (int j = 0; j < 16; ++j) {
        int v = lane * 16 + j;
        OT[(long)(b * FF + v) * NK + m] = f2bf(p[j] * inv);
    }
}

// ---------------------------------------------------------------------------
// Flash attention. grid.x = B*NQ/QT ; block = 256 (8 waves = 2 Q-groups x 4 col-groups)
// Tiles staged with GLOBAL_LOAD_ASYNC_TO_LDS_B128; wk waited before S-phase,
// wv^T wait deferred until after softmax (overlaps WMMA + exp with the DMA).
// LDS ~135 KB (CDNA5: 320 KB/WGP).
// ---------------------------------------------------------------------------
#define QT  32
#define KT  64
#define DSTR 264   // wq/wk LDS row stride (bf16): 256 + 8 pad
#define VSTR 72    // wv  LDS row stride: 64 + 8 pad
#define SSTR 68    // S   LDS row stride (f32)
#define PSTR 72    // P   LDS row stride (bf16)
#define FLASH_SMEM (QT*DSTR*2 + KT*DSTR*2 + FF*VSTR*2 + QT*SSTR*4 + QT*PSTR*2 + 3*QT*4)

__global__ void __launch_bounds__(256, 1)
flash_attn_kernel(const unsigned short* __restrict__ gwq,
                  const unsigned short* __restrict__ gwk,
                  const unsigned short* __restrict__ gwvT,
                  const float* __restrict__ gq,
                  float* __restrict__ gout)
{
    const int t    = threadIdx.x;
    const int lane = t & 31;
    const int wave = t >> 5;
    const int lo   = lane & 15;
    const int hi   = lane >> 4;
    const int qg   = wave >> 2;
    const int cg   = wave & 3;
    const int b    = blockIdx.x >> 6;   // NQ/QT = 64 tiles per batch
    const int qt   = blockIdx.x & 63;

    extern __shared__ char smem[];
    unsigned short* wq_s = (unsigned short*)smem;
    unsigned short* wk_s = wq_s + QT * DSTR;
    unsigned short* wv_s = wk_s + KT * DSTR;
    float*          S_s  = (float*)(wv_s + FF * VSTR);
    unsigned short* P_s  = (unsigned short*)(S_s + QT * SSTR);
    float*          mrow = (float*)(P_s + QT * PSTR);
    float*          lrow = mrow + QT;
    float*          srow = lrow + QT;

    // ---- stage wq tile (32 x 256 bf16) via async DMA ----
    {
        const void* sb = (const void*)(gwq + ((size_t)(b * NQ + qt * QT)) * DD);
#pragma unroll
        for (int i = 0; i < 4; ++i) {
            int idx = t + i * 256;
            int row = idx >> 5, c = idx & 31;               // 32 x 16B per row
            async_ld16((char*)wq_s + row * (DSTR * 2) + c * 16, sb,
                       (unsigned)(row * 512 + c * 16));
        }
    }
    if (t < QT) { mrow[t] = -3.0e38f; lrow[t] = 0.f; }
    wait_async_0();
    __syncthreads();

    // ---- preload the 8 wq A-fragments (reused for all 32 K-tiles) ----
    BF16Frag aq[8];
#pragma unroll
    for (int d = 0; d < 8; ++d) {
        const char* ab = (const char*)wq_s + ((qg * 16 + lo) * DSTR + d * 32 + hi * 8) * 2;
        aq[d].q[0] = *(const uint4*)ab;
        aq[d].q[1] = *(const uint4*)(ab + 32);
    }

    const v8f vzero = {0.f,0.f,0.f,0.f,0.f,0.f,0.f,0.f};
    v8f o[8];
#pragma unroll
    for (int f = 0; f < 8; ++f) o[f] = vzero;

    for (int ktile = 0; ktile < NK / KT; ++ktile) {
        // ---- async: wk tile (64 x 256 bf16 = 32 KB), 8 x 16B per thread ----
        {
            const void* sb = (const void*)(gwk + ((size_t)(b * NK + ktile * KT)) * DD);
#pragma unroll
            for (int i = 0; i < 8; ++i) {
                int idx = t + i * 256;
                int row = idx >> 5, c = idx & 31;
                async_ld16((char*)wk_s + row * (DSTR * 2) + c * 16, sb,
                           (unsigned)(row * 512 + c * 16));
            }
        }
        // ---- async: wv^T tile (512 v-rows x 64 m = 64 KB), 16 x 16B per thread ----
        {
            const void* sb = (const void*)(gwvT + ((size_t)b * FF) * NK);
#pragma unroll
            for (int i = 0; i < 16; ++i) {
                int idx = t + i * 256;
                int v = idx >> 3, c = idx & 7;              // 8 x 16B per v-row
                async_ld16((char*)wv_s + v * (VSTR * 2) + c * 16, sb,
                           (unsigned)((v * NK + ktile * KT) * 2 + c * 16));
            }
        }
        // prefetch next K-tile toward GL2
        if (ktile + 1 < NK / KT) {
            const unsigned short* nk = gwk + ((size_t)(b * NK + (ktile + 1) * KT)) * DD;
            __builtin_prefetch(nk + (size_t)t * 64, 0, 1);
            const unsigned short* nv = gwvT + ((size_t)b * FF) * NK + (ktile + 1) * KT;
            __builtin_prefetch(nv + (size_t)t * 64, 0, 1);
        }
        wait_async_le16();      // first 8 (wk) complete; 16 wv still in flight
        __syncthreads();

        // ---- S = wq * wk^T : each wave one 16x16 chunk ----
        v8f s = vzero;
#pragma unroll
        for (int d = 0; d < 8; ++d) {
            BF16Frag bk;   // lane n=lo -> wk row cg*16+lo, chunks at d*32+hi*8 / +16
            const char* bb = (const char*)wk_s + ((cg * 16 + lo) * DSTR + d * 32 + hi * 8) * 2;
            bk.q[0] = *(const uint4*)bb;
            bk.q[1] = *(const uint4*)(bb + 32);
            s = wmma_bf16(aq[d], bk, s);
        }
        {
            float* sp = S_s + (qg * 16 + hi * 8) * SSTR + cg * 16 + lo;
#pragma unroll
            for (int r = 0; r < 8; ++r) sp[r * SSTR] = s[r];
        }
        __syncthreads();

        // ---- online softmax over the 64 new columns (wave 0, lane per row) ----
        if (t < QT) {
            const float temp = 0.0625f;   // 256^-0.5
            const float4* row = (const float4*)(S_s + t * SSTR);
            float mo = mrow[t];
            float mx = mo;
#pragma unroll
            for (int j = 0; j < 16; ++j) {
                float4 v = row[j];
                mx = fmaxf(mx, fmaxf(fmaxf(v.x * temp, v.y * temp),
                                     fmaxf(v.z * temp, v.w * temp)));
            }
            const float scale = __expf(mo - mx);
            float sum = 0.f;
            uint2* pr = (uint2*)(P_s + t * PSTR);
#pragma unroll
            for (int j = 0; j < 16; ++j) {
                float4 v = row[j];
                float p0 = __expf(v.x * temp - mx);
                float p1 = __expf(v.y * temp - mx);
                float p2 = __expf(v.z * temp - mx);
                float p3 = __expf(v.w * temp - mx);
                sum += (p0 + p1) + (p2 + p3);
                uint2 u; u.x = f2bf2(p0, p1); u.y = f2bf2(p2, p3);
                pr[j] = u;
            }
            mrow[t] = mx;
            lrow[t] = lrow[t] * scale + sum;
            srow[t] = scale;
        }
        wait_async_0();         // wv^T tile now resident in LDS
        __syncthreads();

        // ---- rescale O, then O += P * wv (16 rows x 128 v-cols per wave) ----
        float scl[8];
        {
            const float* sp = srow + qg * 16 + hi * 8;
#pragma unroll
            for (int j = 0; j < 8; ++j) scl[j] = sp[j];
        }
#pragma unroll
        for (int f = 0; f < 8; ++f)
#pragma unroll
            for (int j = 0; j < 8; ++j) o[f][j] *= scl[j];

#pragma unroll
        for (int ks = 0; ks < 2; ++ks) {
            BF16Frag ap;   // A = P rows qg*16+lo, chunks at m = ks*32+hi*8 / +16
            const char* ab = (const char*)P_s + ((qg * 16 + lo) * PSTR + ks * 32 + hi * 8) * 2;
            ap.q[0] = *(const uint4*)ab;
            ap.q[1] = *(const uint4*)(ab + 32);
#pragma unroll
            for (int f = 0; f < 8; ++f) {
                BF16Frag bv;   // lane n=lo -> wv row cg*128+f*16+lo, chunks at ks*32+hi*8 / +16
                const char* bb =
                    (const char*)wv_s + ((cg * 128 + f * 16 + lo) * VSTR + ks * 32 + hi * 8) * 2;
                bv.q[0] = *(const uint4*)bb;
                bv.q[1] = *(const uint4*)(bb + 32);
                o[f] = wmma_bf16(ap, bv, o[f]);
            }
        }
        __syncthreads();
    }

    // ---- epilogue: out = O / l + query ----
    float inv[8];
    {
        const float* lp = lrow + qg * 16 + hi * 8;
#pragma unroll
        for (int j = 0; j < 8; ++j) inv[j] = 1.0f / lp[j];
    }
    const int qbase = b * NQ + qt * QT + qg * 16 + hi * 8;
    const int vbase = cg * 128 + lo;
#pragma unroll
    for (int f = 0; f < 8; ++f)
#pragma unroll
        for (int r = 0; r < 8; ++r) {
            size_t idx = (size_t)(qbase + r) * FF + vbase + f * 16;
            gout[idx] = o[f][r] * inv[r] + gq[idx];
        }
}

// ---------------------------------------------------------------------------
// Host launcher
// ---------------------------------------------------------------------------
extern "C" void kernel_launch(void* const* d_in, const int* in_sizes, int n_in,
                              void* d_out, int out_size, void* d_ws, size_t ws_size,
                              hipStream_t stream)
{
    (void)in_sizes; (void)n_in; (void)out_size; (void)ws_size;

    const float* query = (const float*)d_in[0];
    const float* key   = (const float*)d_in[1];
    const float* value = (const float*)d_in[2];
    const float* WQ    = (const float*)d_in[3];   // [256, 512]
    const float* WK    = (const float*)d_in[4];   // [256, 256]
    const float* WV    = (const float*)d_in[5];   // [512, 256]
    float* out = (float*)d_out;

    // workspace: wq(16MB) | wk(16MB) | wvT(32MB) | f32 staging(64MB)
    unsigned short* wq   = (unsigned short*)d_ws;
    unsigned short* wk   = wq  + (size_t)RR * DD;
    unsigned short* wvT  = wk  + (size_t)RR * DD;
    float*          stag = (float*)(wvT + (size_t)BB * FF * NK);

    hipFuncSetAttribute((const void*)flash_attn_kernel,
                        hipFuncAttributeMaxDynamicSharedMemorySize, FLASH_SMEM);

    dim3 blk(256);

    proj_gemm_kernel<<<dim3(RR / 128, 256 / 64), blk, 0, stream>>>(query, WQ, stag, 512, 256);
    rownorm_kernel<<<RR / 8, blk, 0, stream>>>(stag, wq, 256);

    proj_gemm_kernel<<<dim3(RR / 128, 256 / 64), blk, 0, stream>>>(key, WK, stag, 256, 256);
    rownorm_kernel<<<RR / 8, blk, 0, stream>>>(stag, wk, 256);

    proj_gemm_kernel<<<dim3(RR / 128, 512 / 64), blk, 0, stream>>>(value, WV, stag, 256, 512);
    rownorm_T_kernel<<<RR / 8, blk, 0, stream>>>(stag, wvT);

    flash_attn_kernel<<<BB * (NQ / QT), blk, FLASH_SMEM, stream>>>(wq, wk, wvT, query, out);
}
